// Input_level_Adapeter_27350351741374
// MI455X (gfx1250) — matrix operand order repaired and checked by reading, not compiled, not run on hardware
//
#include <hip/hip_runtime.h>
#include <math.h>

// ---------------- CDNA5 WMMA types ----------------
typedef __attribute__((ext_vector_type(16))) _Float16 v16h;
typedef __attribute__((ext_vector_type(8)))  _Float16 v8h;
typedef __attribute__((ext_vector_type(8)))  float    v8f;
typedef __attribute__((ext_vector_type(4)))  float    v4f;

#define B_N   8
#define H_IN  512
#define W_IN  512
#define HW    (H_IN * W_IN)
#define NTOK  1024          // (512/16)^2 tokens after 4 stride-2 convs
#define DIMC  64

__device__ __forceinline__ float fast_tanh(float x)
{
    x = fminf(fmaxf(x, -10.0f), 10.0f);
    float e = __expf(2.0f * x);
    return (e - 1.0f) / (e + 1.0f);
}

// =====================================================================
// Stride-2 3x3 conv + eval-BN (+ optional exact GELU).  Small fraction of
// total FLOPs -> simple VALU kernel, one output element per thread.
// tok_major=1 writes (B, N, Cout) so the WMMA stage gets coalesced A rows.
// =====================================================================
__global__ void conv_down_kernel(const float* __restrict__ in, const float* __restrict__ w,
                                 const float* __restrict__ bias, float* __restrict__ out,
                                 int Cin, int Cout, int Hin, int Win, int apply_gelu,
                                 int tok_major)
{
    const int Hout = Hin >> 1, Wout = Win >> 1;
    int idx = blockIdx.x * blockDim.x + threadIdx.x;
    int total = B_N * Cout * Hout * Wout;
    if (idx >= total) return;
    int wo = idx % Wout;
    int t  = idx / Wout;
    int ho = t % Hout; t /= Hout;
    int co = t % Cout;
    int b  = t / Cout;
    const float* wp = w + (size_t)co * Cin * 9;
    float acc = bias[co];
    int hi0 = 2 * ho - 1, wi0 = 2 * wo - 1;
    for (int ci = 0; ci < Cin; ++ci) {
        const float* ip = in + (((size_t)b * Cin + ci) * Hin) * Win;
        __builtin_prefetch(ip + (size_t)(hi0 > 0 ? hi0 : 0) * Win, 0, 1);
        const float* wc = wp + ci * 9;
        #pragma unroll
        for (int kh = 0; kh < 3; ++kh) {
            int hi = hi0 + kh;
            if ((unsigned)hi >= (unsigned)Hin) continue;
            const float* row = ip + (size_t)hi * Win;
            #pragma unroll
            for (int kw = 0; kw < 3; ++kw) {
                int wi = wi0 + kw;
                if ((unsigned)wi < (unsigned)Win)
                    acc = fmaf(row[wi], wc[kh * 3 + kw], acc);
            }
        }
    }
    acc *= 0.9999950000374997f;                 // 1/sqrt(1 + 1e-5) eval BN
    if (apply_gelu)
        acc = 0.5f * acc * (1.0f + erff(acc * 0.70710678118654752f));
    if (tok_major)
        out[((size_t)b * Hout * Wout + (size_t)ho * Wout + wo) * Cout + co] = acc;
    else
        out[idx] = acc;
}

// =====================================================================
// K/V projections as WMMA GEMM: k = dx @ kw^T + kb  (dx: B x 1024 x 64,
// token-major).  Block = 8 waves; waves 0-3 -> K proj, 4-7 -> V proj.
// K=64 as two accumulating 16x16x32 f16 WMMAs; fragments built from
// coalesced 128-bit loads (each lane's K run is 16 contiguous).
// =====================================================================
__global__ void __launch_bounds__(256)
attn_kv_wmma_kernel(const float* __restrict__ dx,            // (B, 1024, 64)
                    const float* __restrict__ kw_, const float* __restrict__ kb_,
                    const float* __restrict__ vw_, const float* __restrict__ vb_,
                    float* __restrict__ kout, float* __restrict__ vout)  // (B,1024,64)
{
    int blk   = blockIdx.x;
    int ntile = blk % (NTOK / 16);
    int b     = blk / (NTOK / 16);
    int wv    = threadIdx.x >> 5;
    int lane  = threadIdx.x & 31;
    int ctile = wv & 3;
    int isv   = wv >> 2;
    const float* wmat = isv ? vw_ : kw_;
    const float* bvec = isv ? vb_ : kb_;
    float*       obuf = isv ? vout : kout;

    int m  = lane & 15;
    int n  = ntile * 16 + m;
    int hi = lane >> 4;                 // lane half select per ISA layout
    int nn = lane & 15;

    v8f acc = {};
    for (int kstep = 0; kstep < 2; ++kstep) {
        // A fragment: lane holds rows m, K = kbase+hi*16+{0..15}
        const float* pA = dx + ((size_t)b * NTOK + n) * DIMC + kstep * 32 + hi * 16;
        v4f a0 = *(const v4f*)(pA);
        v4f a1 = *(const v4f*)(pA + 4);
        v4f a2 = *(const v4f*)(pA + 8);
        v4f a3 = *(const v4f*)(pA + 12);
        v16h a;
        #pragma unroll
        for (int i = 0; i < 4; ++i) {
            a[i]      = (_Float16)a0[i];
            a[4 + i]  = (_Float16)a1[i];
            a[8 + i]  = (_Float16)a2[i];
            a[12 + i] = (_Float16)a3[i];
        }
        // B fragment: B[k][n'] = w[n', k]; per-lane K run is 16 contiguous
        const float* pB = wmat + ((size_t)ctile * 16 + nn) * DIMC + kstep * 32 + hi * 16;
        v4f b0 = *(const v4f*)(pB);
        v4f b1 = *(const v4f*)(pB + 4);
        v4f b2 = *(const v4f*)(pB + 8);
        v4f b3 = *(const v4f*)(pB + 12);
        v16h bf;
        #pragma unroll
        for (int i = 0; i < 4; ++i) {
            bf[i]      = (_Float16)b0[i];
            bf[4 + i]  = (_Float16)b1[i];
            bf[8 + i]  = (_Float16)b2[i];
            bf[12 + i] = (_Float16)b3[i];
        }
        acc = __builtin_amdgcn_wmma_f32_16x16x32_f16(false, a, false, bf,
                                                     (short)0, acc, false, false);
    }
    int   col    = lane & 15;
    int   cprime = ctile * 16 + col;
    float bb     = bvec[cprime];
    int   rbase  = hi * 8;
    #pragma unroll
    for (int r = 0; r < 8; ++r) {
        int nrow = ntile * 16 + r + rbase;
        obuf[((size_t)b * NTOK + nrow) * DIMC + cprime] = acc[r] + bb;
    }
}

// =====================================================================
// Softmax attention tail: one block per (b, t).
// =====================================================================
__global__ void attn_finish_kernel(const float* __restrict__ kbuf, const float* __restrict__ vbuf,
                                   const float* __restrict__ q,   // (T, 64)
                                   const float* __restrict__ pw, const float* __restrict__ pb,
                                   const float* __restrict__ dw, const float* __restrict__ db,
                                   float* __restrict__ o_out, int T)
{
    __shared__ float sc[NTOK];
    __shared__ float red[256];
    __shared__ float ovec[DIMC];
    int b   = blockIdx.x / T;
    int t   = blockIdx.x % T;
    int tid = threadIdx.x;
    const float scale = 0.125f;                       // 64^-0.5
    const float* qt = q + (size_t)t * DIMC;

    for (int i = tid; i < NTOK; i += 256) {
        const float* kr = kbuf + ((size_t)b * NTOK + i) * DIMC;
        float d = 0.f;
        for (int c = 0; c < DIMC; ++c) d = fmaf(qt[c], kr[c], d);
        sc[i] = d * scale;
    }
    __syncthreads();
    float mx = -1e30f;
    for (int i = tid; i < NTOK; i += 256) mx = fmaxf(mx, sc[i]);
    red[tid] = mx; __syncthreads();
    for (int s = 128; s > 0; s >>= 1) {
        if (tid < s) red[tid] = fmaxf(red[tid], red[tid + s]);
        __syncthreads();
    }
    mx = red[0]; __syncthreads();
    float psum = 0.f;
    for (int i = tid; i < NTOK; i += 256) { float e = __expf(sc[i] - mx); sc[i] = e; psum += e; }
    red[tid] = psum; __syncthreads();
    for (int s = 128; s > 0; s >>= 1) {
        if (tid < s) red[tid] += red[tid + s];
        __syncthreads();
    }
    float inv = 1.0f / red[0];
    __syncthreads();
    if (tid < DIMC) {
        float a = 0.f;
        for (int i = 0; i < NTOK; ++i)
            a = fmaf(sc[i], vbuf[((size_t)b * NTOK + i) * DIMC + tid], a);
        ovec[tid] = a * inv;
    }
    __syncthreads();
    float part = 0.f;
    if (tid < DIMC) {
        float pr = pb[tid];
        for (int j = 0; j < DIMC; ++j) pr = fmaf(pw[tid * DIMC + j], ovec[j], pr);
        part = dw[tid] * pr;
    }
    red[tid] = part; __syncthreads();
    for (int s = 128; s > 0; s >>= 1) {
        if (tid < s) red[tid] += red[tid + s];
        __syncthreads();
    }
    if (tid == 0) o_out[b * T + t] = red[0] + db[0];
}

// =====================================================================
// Per-image gain + gaussian unsharp (reflect pad), clip -> I2.
// =====================================================================
__global__ void gain_denoise_kernel(const float* __restrict__ I1, const float* __restrict__ okp,
                                    const float* __restrict__ gain_base, float* __restrict__ I2)
{
    size_t idx = (size_t)blockIdx.x * blockDim.x + threadIdx.x;
    const size_t total = (size_t)B_N * 3 * HW;
    if (idx >= total) return;
    int wq = (int)(idx % W_IN); size_t t = idx / W_IN;
    int hq = (int)(t % H_IN);   t /= H_IN;
    int c  = (int)(t % 3);
    int b  = (int)(t / 3);
    const float* o = okp + b * 4;
    float r1   = 0.1f * o[0] + 3.0f;
    float r2   = 0.1f * o[1] + 2.0f;
    float gain = o[2] + gain_base[0];
    float sig  = 1.0f / (1.0f + __expf(-o[3]));
    float ex  = __expf(-0.5f / (r1 * r1));
    float gx0 = ex / (2.f * ex + 1.f), gx1 = 1.f / (2.f * ex + 1.f);
    float ey  = __expf(-0.5f / (r2 * r2));
    float gy0 = ey / (2.f * ey + 1.f), gy1 = 1.f / (2.f * ey + 1.f);
    float gx[3] = {gx0, gx1, gx0};
    float gy[3] = {gy0, gy1, gy0};
    const float* plane = I1 + ((size_t)b * 3 + c) * HW;
    float blur = 0.f;
    #pragma unroll
    for (int i = 0; i < 3; ++i) {
        int h = hq + i - 1; h = h < 0 ? -h : (h >= H_IN ? 2 * H_IN - 2 - h : h);
        #pragma unroll
        for (int j = 0; j < 3; ++j) {
            int w = wq + j - 1; w = w < 0 ? -w : (w >= W_IN ? 2 * W_IN - 2 - w : w);
            blur = fmaf(gy[i] * gx[j], plane[(size_t)h * W_IN + w], blur);
        }
    }
    blur *= gain;                                  // conv is linear in the gain
    float center = plane[(size_t)hq * W_IN + wq];
    float outv = blur + sig * (center - blur);
    I2[idx] = fminf(fmaxf(outv, 1e-5f), 1.0f);
}

// =====================================================================
// Tiny per-image param kernels.
// =====================================================================
__global__ void ccm_kernel(const float* __restrict__ omp, float* __restrict__ ccm,
                           float* __restrict__ dist)
{
    int b = threadIdx.x;
    if (b >= B_N) return;
    const float* o = omp + b * 10;
    for (int i = 0; i < 9; ++i) {
        float v = 0.1f * o[i];
        if (i == 0 || i == 4 || i == 8) v += 1.0f;
        ccm[b * 9 + i] = v;
    }
    dist[b] = fmaxf(o[9], 0.f) + 1.0f;
}

__global__ void zero_kernel(float* p, int n)
{
    int i = blockIdx.x * blockDim.x + threadIdx.x;
    if (i < n) p[i] = 0.f;
}

// Shades-of-gray partial sums: sum_{h,w} I2^dist per (b, c).
__global__ void sog_reduce_kernel(const float* __restrict__ I2, const float* __restrict__ dist,
                                  float* __restrict__ chsum)
{
    __shared__ float red[256];
    const int NB = 64;
    int plane = blockIdx.x / NB;       // b*3 + c
    int blk   = blockIdx.x % NB;
    int b     = plane / 3;
    float d   = dist[b];
    const float* p = I2 + (size_t)plane * HW;
    int start = blk * (HW / NB);
    float s = 0.f;
    for (int i = threadIdx.x; i < HW / NB; i += 256)
        s += __powf(p[start + i], d);
    red[threadIdx.x] = s; __syncthreads();
    for (int st = 128; st > 0; st >>= 1) {
        if (threadIdx.x < st) red[threadIdx.x] += red[threadIdx.x + st];
        __syncthreads();
    }
    if (threadIdx.x == 0) atomicAdd(&chsum[plane], red[0]);
}

__global__ void wb_finalize_kernel(const float* __restrict__ chsum, const float* __restrict__ dist,
                                   float* __restrict__ wbscale)
{
    int i = threadIdx.x;
    if (i >= B_N * 3) return;
    int b = i / 3;
    float invd = 1.0f / dist[b];
    float ch   = __powf(chsum[i] / (float)HW, invd);
    float tot  = (chsum[b * 3] + chsum[b * 3 + 1] + chsum[b * 3 + 2]) / (3.0f * HW);
    float avg  = __powf(tot, invd);
    wbscale[i] = avg / ch;                          // I3 = I2 * avg / ch
}

// =====================================================================
// Pre-pack NILUT weights into WMMA B-fragment layout (f16), once.
// wfrag layout: [layer(5)][ntile(2)][lane(32)][h(16)] halfs.
// =====================================================================
__global__ void prep_lut_weights_kernel(const float* __restrict__ w0, const float* __restrict__ w1,
                                        const float* __restrict__ w2, const float* __restrict__ w3,
                                        const float* __restrict__ w4, _Float16* __restrict__ wfrag)
{
    int tid = threadIdx.x;              // 0..319
    if (tid >= 320) return;
    int lane = tid & 31;
    int nt   = (tid >> 5) & 1;
    int l    = tid / 64;
    const float* W; int K, N;
    switch (l) {
        case 0:  W = w0; K = 3;  N = 32; break;
        case 1:  W = w1; K = 32; N = 32; break;
        case 2:  W = w2; K = 32; N = 32; break;
        case 3:  W = w3; K = 32; N = 32; break;
        default: W = w4; K = 32; N = 3;  break;
    }
    int nn    = lane & 15;
    int col   = nt * 16 + nn;
    int kbase = (lane < 16) ? 0 : 16;
    _Float16* dst = wfrag + (((size_t)l * 2 + nt) * 32 + lane) * 16;
    for (int h = 0; h < 16; ++h) {
        int kk = (h < 8 ? h : h + 8) + (kbase >> 1);
        float v = (kk < K && col < N) ? W[col * K + kk] : 0.0f;
        dst[h] = (_Float16)v;
    }
}

// =====================================================================
// Fused WB + CCM + NILUT MLP (the FLOP-dominant stage).
// 256 threads = 8 waves; each wave owns a 16-pixel tile end-to-end, so
// ALL LDS traffic is wave-private: per-wave in-order DScnt ordering
// (s_wait_dscnt) replaces block barriers.  Activations staged as f16 so
// A-fragments are two 128-bit ds loads per lane.
// =====================================================================
__global__ void __launch_bounds__(256)
wb_ccm_nilut_kernel(const float* __restrict__ I2, const float* __restrict__ wbscale,
                    const float* __restrict__ ccm, const _Float16* __restrict__ wfrag,
                    const float* __restrict__ lb0, const float* __restrict__ lb1,
                    const float* __restrict__ lb2, const float* __restrict__ lb3,
                    const float* __restrict__ lb4, float* __restrict__ out)
{
    __shared__ alignas(16) _Float16 xh[8][16][32];   // per-wave f16 activation tiles
    __shared__ float i4buf[8][16][4];                // residual input
    int tid  = threadIdx.x;
    int wv   = tid >> 5;
    int lane = tid & 31;
    int m    = lane & 15;
    int hi   = lane >> 4;

    // Stage 1 (wave-local): lanes 0..15 compute WB+CCM -> I4 for pixel m
    if (lane < 16) {
        size_t p  = (size_t)blockIdx.x * 128 + wv * 16 + lane;
        int b  = (int)(p / HW);
        int hw = (int)(p % HW);
        const float* base = I2 + (size_t)b * 3 * HW + hw;
        float i3[3];
        #pragma unroll
        for (int c = 0; c < 3; ++c)
            i3[c] = base[(size_t)c * HW] * wbscale[b * 3 + c];
        const float* M = ccm + b * 9;
        #pragma unroll
        for (int d = 0; d < 3; ++d) {
            float v = M[d * 3 + 0] * i3[0] + M[d * 3 + 1] * i3[1] + M[d * 3 + 2] * i3[2];
            v = fminf(fmaxf(v, 1e-5f), 1.0f);
            i4buf[wv][lane][d] = v;
            xh[wv][lane][d]    = (_Float16)v;
        }
        #pragma unroll
        for (int c = 3; c < 32; ++c) xh[wv][lane][c] = (_Float16)0.0f;  // K-pad
    }
    __asm__ volatile("s_wait_dscnt 0x0" ::: "memory");   // wave-local LDS ordering

    const float* biases[4] = {lb0, lb1, lb2, lb3};

    #pragma unroll
    for (int l = 0; l < 4; ++l) {
        // A fragment: per-lane halves = two contiguous 8-half runs of row m
        v8h lo  = *(const v8h*)(&xh[wv][m][hi * 8]);
        v8h hi8 = *(const v8h*)(&xh[wv][m][hi * 8 + 16]);
        v16h a;
        #pragma unroll
        for (int i = 0; i < 8; ++i) { a[i] = lo[i]; a[8 + i] = hi8[i]; }

        v16h bf0 = *(const v16h*)(wfrag + (((size_t)l * 2 + 0) * 32 + lane) * 16);
        v16h bf1 = *(const v16h*)(wfrag + (((size_t)l * 2 + 1) * 32 + lane) * 16);
        v8f c0 = {}, c1 = {};
        c0 = __builtin_amdgcn_wmma_f32_16x16x32_f16(false, a, false, bf0, (short)0, c0, false, false);
        c1 = __builtin_amdgcn_wmma_f32_16x16x32_f16(false, a, false, bf1, (short)0, c1, false, false);

        // Write back activations (loads above precede these stores in the
        // wave's in-order LDS stream, so no hazard on the overwrite).
        int col   = lane & 15;
        int rbase = hi * 8;
        float bb0 = biases[l][col], bb1 = biases[l][16 + col];
        #pragma unroll
        for (int r = 0; r < 8; ++r) {
            float v0 = c0[r] + bb0;
            float v1 = c1[r] + bb1;
            if (l == 0) { v0 = fmaxf(v0, 0.f);  v1 = fmaxf(v1, 0.f); }
            else        { v0 = fast_tanh(v0);   v1 = fast_tanh(v1); }
            xh[wv][r + rbase][col]      = (_Float16)v0;
            xh[wv][r + rbase][16 + col] = (_Float16)v1;
        }
        __asm__ volatile("s_wait_dscnt 0x0" ::: "memory");
    }

    // Layer 4: 32 -> 3, + residual I4, clip [0,1], write (B,3,H,W)
    {
        v8h lo  = *(const v8h*)(&xh[wv][m][hi * 8]);
        v8h hi8 = *(const v8h*)(&xh[wv][m][hi * 8 + 16]);
        v16h a;
        #pragma unroll
        for (int i = 0; i < 8; ++i) { a[i] = lo[i]; a[8 + i] = hi8[i]; }
        v16h bf = *(const v16h*)(wfrag + (((size_t)4 * 2 + 0) * 32 + lane) * 16);
        v8f c = {};
        c = __builtin_amdgcn_wmma_f32_16x16x32_f16(false, a, false, bf, (short)0, c, false, false);
        int col   = lane & 15;
        int rbase = hi * 8;
        if (col < 3) {
            float bb = lb4[col];
            #pragma unroll
            for (int r = 0; r < 8; ++r) {
                int mm = r + rbase;
                float v = c[r] + bb + i4buf[wv][mm][col];
                v = fminf(fmaxf(v, 0.f), 1.0f);
                size_t p = (size_t)blockIdx.x * 128 + wv * 16 + mm;
                int b  = (int)(p / HW);
                int hw = (int)(p % HW);
                out[((size_t)b * 3 + col) * HW + hw] = v;
            }
        }
    }
}

// =====================================================================
// Host side
// =====================================================================
extern "C" void kernel_launch(void* const* d_in, const int* in_sizes, int n_in,
                              void* d_out, int out_size, void* d_ws, size_t ws_size,
                              hipStream_t stream)
{
    (void)in_sizes; (void)n_in; (void)out_size; (void)ws_size;

    const float* I1 = (const float*)d_in[0];
    const float* kp_q = (const float*)d_in[1];
    const float* kp_cw[4] = {(const float*)d_in[2], (const float*)d_in[4],
                             (const float*)d_in[6], (const float*)d_in[8]};
    const float* kp_cb[4] = {(const float*)d_in[3], (const float*)d_in[5],
                             (const float*)d_in[7], (const float*)d_in[9]};
    const float* kp_kw = (const float*)d_in[10];
    const float* kp_vw = (const float*)d_in[11];
    const float* kp_pw = (const float*)d_in[12];
    const float* kp_dw = (const float*)d_in[13];
    const float* kp_kb = (const float*)d_in[14];
    const float* kp_vb = (const float*)d_in[15];
    const float* kp_pb = (const float*)d_in[16];
    const float* kp_db = (const float*)d_in[17];
    const float* mp_q = (const float*)d_in[18];
    const float* mp_cw[4] = {(const float*)d_in[19], (const float*)d_in[21],
                             (const float*)d_in[23], (const float*)d_in[25]};
    const float* mp_cb[4] = {(const float*)d_in[20], (const float*)d_in[22],
                             (const float*)d_in[24], (const float*)d_in[26]};
    const float* mp_kw = (const float*)d_in[27];
    const float* mp_vw = (const float*)d_in[28];
    const float* mp_pw = (const float*)d_in[29];
    const float* mp_dw = (const float*)d_in[30];
    const float* mp_kb = (const float*)d_in[31];
    const float* mp_vb = (const float*)d_in[32];
    const float* mp_pb = (const float*)d_in[33];
    const float* mp_db = (const float*)d_in[34];
    const float* gain_base = (const float*)d_in[35];
    const float* lw[5] = {(const float*)d_in[36], (const float*)d_in[38],
                          (const float*)d_in[40], (const float*)d_in[42],
                          (const float*)d_in[44]};
    const float* lb[5] = {(const float*)d_in[37], (const float*)d_in[39],
                          (const float*)d_in[41], (const float*)d_in[43],
                          (const float*)d_in[45]};

    // ---- workspace layout (256-B aligned slices) ----
    char* ws = (char*)d_ws;
    size_t off = 0;
    #define WS_TAKE(ty, name, bytes) \
        ty* name = (ty*)(ws + off);  \
        off = (off + (size_t)(bytes) + 255) & ~(size_t)255;
    WS_TAKE(float, s1,    (size_t)B_N *  8 * 256 * 256 * 4)   // 16.8 MB
    WS_TAKE(float, s2,    (size_t)B_N * 16 * 128 * 128 * 4)
    WS_TAKE(float, s3,    (size_t)B_N * 32 *  64 *  64 * 4)
    WS_TAKE(float, s4,    (size_t)B_N * 64 *  32 *  32 * 4)   // token-major (B,N,64)
    WS_TAKE(float, I2f,   (size_t)B_N *  3 * HW * 4)          // 25.2 MB
    WS_TAKE(float, kbuf,  (size_t)B_N * NTOK * DIMC * 4)
    WS_TAKE(float, vbuf,  (size_t)B_N * NTOK * DIMC * 4)
    WS_TAKE(float, o_kp,  B_N * 4 * 4)
    WS_TAKE(float, o_mp,  B_N * 10 * 4)
    WS_TAKE(float, ccmb,  B_N * 9 * 4)
    WS_TAKE(float, distb, B_N * 4)
    WS_TAKE(float, chsum, B_N * 3 * 4)
    WS_TAKE(float, wbs,   B_N * 3 * 4)
    WS_TAKE(_Float16, wfragh, 5 * 2 * 32 * 16 * 2)
    #undef WS_TAKE

    auto run_downsample = [&](const float* img, const float* const* cw, const float* const* cb) {
        conv_down_kernel<<<16384, 256, 0, stream>>>(img, cw[0], cb[0], s1,  3,  8, 512, 512, 1, 0);
        conv_down_kernel<<< 8192, 256, 0, stream>>>(s1,  cw[1], cb[1], s2,  8, 16, 256, 256, 1, 0);
        conv_down_kernel<<< 4096, 256, 0, stream>>>(s2,  cw[2], cb[2], s3, 16, 32, 128, 128, 1, 0);
        conv_down_kernel<<< 2048, 256, 0, stream>>>(s3,  cw[3], cb[3], s4, 32, 64,  64,  64, 0, 1);
    };

    // Pre-pack NILUT weights into WMMA B-fragment layout
    prep_lut_weights_kernel<<<1, 320, 0, stream>>>(lw[0], lw[1], lw[2], lw[3], lw[4], wfragh);

    // ---- Kernel_Predictor ----
    run_downsample(I1, kp_cw, kp_cb);
    attn_kv_wmma_kernel<<<B_N * (NTOK / 16), 256, 0, stream>>>(s4, kp_kw, kp_kb, kp_vw, kp_vb, kbuf, vbuf);
    attn_finish_kernel<<<B_N * 4, 256, 0, stream>>>(kbuf, vbuf, kp_q, kp_pw, kp_pb, kp_dw, kp_db, o_kp, 4);

    // ---- gain + gaussian unsharp denoise ----
    gain_denoise_kernel<<<(B_N * 3 * HW) / 256, 256, 0, stream>>>(I1, o_kp, gain_base, I2f);

    // ---- Matrix_Predictor ----
    run_downsample(I2f, mp_cw, mp_cb);
    attn_kv_wmma_kernel<<<B_N * (NTOK / 16), 256, 0, stream>>>(s4, mp_kw, mp_kb, mp_vw, mp_vb, kbuf, vbuf);
    attn_finish_kernel<<<B_N * 10, 256, 0, stream>>>(kbuf, vbuf, mp_q, mp_pw, mp_pb, mp_dw, mp_db, o_mp, 10);
    ccm_kernel<<<1, 32, 0, stream>>>(o_mp, ccmb, distb);

    // ---- Shades-of-Gray stats ----
    zero_kernel<<<1, 32, 0, stream>>>(chsum, B_N * 3);
    sog_reduce_kernel<<<B_N * 3 * 64, 256, 0, stream>>>(I2f, distb, chsum);
    wb_finalize_kernel<<<1, 32, 0, stream>>>(chsum, distb, wbs);

    // ---- fused WB + CCM + NILUT (WMMA) ----
    wb_ccm_nilut_kernel<<<(B_N * HW) / 128, 256, 0, stream>>>(
        I2f, wbs, ccmb, wfragh, lb[0], lb[1], lb[2], lb[3], lb[4], (float*)d_out);
}
